// NeuralImageFunction_68728066670742
// MI455X (gfx1250) — matrix operand order repaired and checked by reading, not compiled, not run on hardware
//
#include <hip/hip_runtime.h>
#include <math.h>

typedef __attribute__((ext_vector_type(16))) _Float16 v16h;
typedef __attribute__((ext_vector_type(8)))  _Float16 v8h;
typedef __attribute__((ext_vector_type(8)))  float    v8f;

#define RES    2048
#define RANKS  128
#define KSZ    101
#define KHALF  50
#define NB     8
#define NPTS   262144

// ---------------------------------------------------------------------------
// Kernel 1: depthwise Gaussian blur of one row (length 2048, 101 taps, SAME/0-pad)
// Both outputs stored TRANSPOSED, K(=rank)-contiguous:
//   tensor 0 (rank1) -> Bt[c][w][r]  (f16)   GEMM B operand
//   tensor 1 (rank2) -> At[c][h][r]  (f16)   GEMM A operand
// ---------------------------------------------------------------------------
__global__ void blur_kernel(const float* __restrict__ rank1,
                            const float* __restrict__ rank2,
                            const float* __restrict__ sigma_p,
                            _Float16* __restrict__ Bt,
                            _Float16* __restrict__ At)
{
    __shared__ float wts[KSZ];
    __shared__ float row[RES];

    const int blk = blockIdx.x;          // 0..767 : {tensor, c, r}
    const int t   = blk / 384;
    const int rem = blk - t * 384;
    const int c   = rem >> 7;            // 0..2
    const int r   = rem & 127;           // 0..127

    const float sg  = sigma_p[0];
    const float s   = fmaxf(sg, 0.01f);
    const float amp = 1.0f / fmaxf(sg * 2.5066282746310002f, 1.0f); // sqrt(2*pi)
    for (int i = threadIdx.x; i < KSZ; i += blockDim.x) {
        const float u = (float)(i - KHALF) / s;
        wts[i] = amp * expf(-0.5f * u * u);
    }

    const float* src = (t == 0 ? rank1 : rank2) + ((size_t)c * RANKS + r) * RES;
    for (int i = threadIdx.x; i < RES; i += blockDim.x) row[i] = src[i];
    __syncthreads();

    _Float16* dst = (t == 0 ? Bt : At);
    for (int i = threadIdx.x; i < RES; i += blockDim.x) {
        float acc = 0.0f;
        for (int j = 0; j < KSZ; ++j) {
            const int p = i + j - KHALF;
            const float v = (p >= 0 && p < RES) ? row[p] : 0.0f;
            acc = fmaf(v, wts[j], acc);
        }
        dst[((size_t)c * RES + i) * RANKS + r] = (_Float16)acc;
    }
}

// ---------------------------------------------------------------------------
// Kernel 2: per-channel GEMM  img[c][h][w] = sum_r At[c][h][r] * Bt[c][w][r]
// One wave32 computes a 32x32 f32 tile: 2 A-frags x 2 B-frags x 4 acc,
// K = 128 in 4 steps of 32 -> 16x v_wmma_f32_16x16x32_f16 (static).
// K-contiguous layouts make every fragment = 2x global_load_b128 per lane:
//   A 16x32 (lane m=lane&15, h=lane>>4): K runs [8h..8h+7], [16+8h..16+8h+7]
//   B 32x16 (lane n=lane&15, h=lane>>4): K runs [8h..8h+7], [16+8h..16+8h+7]
//   D 16x16: row = vgpr + 8h, col = lane&15
// ---------------------------------------------------------------------------
__global__ void gemm_kernel(const _Float16* __restrict__ At,  // [3][2048][128]
                            const _Float16* __restrict__ Bt,  // [3][2048][128]
                            float* __restrict__ img)          // [3][2048][2048]
{
    const int lane = threadIdx.x & 31;
    const int wid  = (blockIdx.x * blockDim.x + threadIdx.x) >> 5; // 0..12287
    const int c    = wid / (64 * 64);
    const int rem  = wid - c * (64 * 64);
    const int tm2  = rem >> 6;    // 32-row tile index, 0..63
    const int tn2  = rem & 63;    // 32-col tile index, 0..63

    const int m = lane & 15;
    const int h = lane >> 4;

    const _Float16* Arow0 = At + ((size_t)c * RES + (size_t)(tm2 * 32 +      m)) * RANKS;
    const _Float16* Arow1 = At + ((size_t)c * RES + (size_t)(tm2 * 32 + 16 + m)) * RANKS;
    const _Float16* Brow0 = Bt + ((size_t)c * RES + (size_t)(tn2 * 32 +      m)) * RANKS;
    const _Float16* Brow1 = Bt + ((size_t)c * RES + (size_t)(tn2 * 32 + 16 + m)) * RANKS;

    union frag { v16h v; v8h half8[2]; };
    frag a0, a1, b0, b1;
    v8f acc00 = {}, acc01 = {}, acc10 = {}, acc11 = {};

#pragma unroll
    for (int k0 = 0; k0 < RANKS; k0 += 32) {
        // A fragments: two contiguous 8-half (16B) runs per lane
        a0.half8[0] = *(const v8h*)(Arow0 + k0 +      8 * h);
        a0.half8[1] = *(const v8h*)(Arow0 + k0 + 16 + 8 * h);
        a1.half8[0] = *(const v8h*)(Arow1 + k0 +      8 * h);
        a1.half8[1] = *(const v8h*)(Arow1 + k0 + 16 + 8 * h);
        // B fragments: two contiguous 8-half (16B) runs per lane
        b0.half8[0] = *(const v8h*)(Brow0 + k0 +      8 * h);
        b0.half8[1] = *(const v8h*)(Brow0 + k0 + 16 + 8 * h);
        b1.half8[0] = *(const v8h*)(Brow1 + k0 +      8 * h);
        b1.half8[1] = *(const v8h*)(Brow1 + k0 + 16 + 8 * h);

        acc00 = __builtin_amdgcn_wmma_f32_16x16x32_f16(false, a0.v, false, b0.v,
                                                       (short)0, acc00, false, false);
        acc01 = __builtin_amdgcn_wmma_f32_16x16x32_f16(false, a0.v, false, b1.v,
                                                       (short)0, acc01, false, false);
        acc10 = __builtin_amdgcn_wmma_f32_16x16x32_f16(false, a1.v, false, b0.v,
                                                       (short)0, acc10, false, false);
        acc11 = __builtin_amdgcn_wmma_f32_16x16x32_f16(false, a1.v, false, b1.v,
                                                       (short)0, acc11, false, false);
    }

    // Store 4 tiles of 16x16
    const size_t row0 = (size_t)c * RES + (size_t)(tm2 * 32);
    float* d00 = img + (row0     ) * RES + (size_t)(tn2 * 32 +      m);
    float* d01 = img + (row0     ) * RES + (size_t)(tn2 * 32 + 16 + m);
    float* d10 = img + (row0 + 16) * RES + (size_t)(tn2 * 32 +      m);
    float* d11 = img + (row0 + 16) * RES + (size_t)(tn2 * 32 + 16 + m);
#pragma unroll
    for (int j = 0; j < 8; ++j) {
        const size_t roff = (size_t)(j + 8 * h) * RES;
        d00[roff] = acc00[j];
        d01[roff] = acc01[j];
        d10[roff] = acc10[j];
        d11[roff] = acc11[j];
    }
}

// ---------------------------------------------------------------------------
// Kernel 3: bilinear grid sample, out[(b*N+n)*3 + c], zero outside image
// ---------------------------------------------------------------------------
__global__ void sample_kernel(const float* __restrict__ img,
                              const float* __restrict__ coord,
                              float* __restrict__ out, int total)
{
    const int i = blockIdx.x * blockDim.x + threadIdx.x;
    if (i >= total) return;

    const float gx = coord[(size_t)i * 2 + 0];
    const float gy = coord[(size_t)i * 2 + 1];
    const float x  = (gx + 1.0f) * (RES * 0.5f) - 0.5f;
    const float y  = (gy + 1.0f) * (RES * 0.5f) - 0.5f;
    const float x0 = floorf(x), y0 = floorf(y);
    const float wx1 = x - x0,   wy1 = y - y0;

    const int xi0 = (int)x0, yi0 = (int)y0;
    const int xi1 = xi0 + 1, yi1 = yi0 + 1;

    const float vx0 = (xi0 >= 0 && xi0 < RES) ? 1.0f : 0.0f;
    const float vx1 = (xi1 >= 0 && xi1 < RES) ? 1.0f : 0.0f;
    const float vy0 = (yi0 >= 0 && yi0 < RES) ? 1.0f : 0.0f;
    const float vy1 = (yi1 >= 0 && yi1 < RES) ? 1.0f : 0.0f;

    const float w00 = (1.0f - wx1) * (1.0f - wy1) * vx0 * vy0;
    const float w10 = wx1 * (1.0f - wy1) * vx1 * vy0;
    const float w01 = (1.0f - wx1) * wy1 * vx0 * vy1;
    const float w11 = wx1 * wy1 * vx1 * vy1;

    const int xc0 = min(max(xi0, 0), RES - 1);
    const int xc1 = min(max(xi1, 0), RES - 1);
    const int yc0 = min(max(yi0, 0), RES - 1);
    const int yc1 = min(max(yi1, 0), RES - 1);

#pragma unroll
    for (int c = 0; c < 3; ++c) {
        const float* base = img + (size_t)c * RES * RES;
        const float v = w00 * base[(size_t)yc0 * RES + xc0]
                      + w10 * base[(size_t)yc0 * RES + xc1]
                      + w01 * base[(size_t)yc1 * RES + xc0]
                      + w11 * base[(size_t)yc1 * RES + xc1];
        out[(size_t)i * 3 + c] = v;
    }
}

// ---------------------------------------------------------------------------
extern "C" void kernel_launch(void* const* d_in, const int* in_sizes, int n_in,
                              void* d_out, int out_size, void* d_ws, size_t ws_size,
                              hipStream_t stream)
{
    const float* rank1 = (const float*)d_in[0];  // (3,128,2048)
    const float* rank2 = (const float*)d_in[1];  // (3,128,2048)
    const float* sigma = (const float*)d_in[2];  // scalar
    const float* coord = (const float*)d_in[3];  // (8,262144,2)
    float* out = (float*)d_out;                  // (8,262144,3)

    char* ws = (char*)d_ws;
    const size_t szBlur = (size_t)3 * RES * RANKS * sizeof(_Float16); // 1.5 MB
    _Float16* Bt  = (_Float16*)ws;                  // r1 blurred [c][w][r]
    _Float16* At  = (_Float16*)(ws + szBlur);       // r2 blurred [c][h][r]
    float*    img = (float*)(ws + 2 * szBlur);      // [3][2048][2048] f32, 48 MB

    // 1) blur: one block per (tensor, channel, rank) row
    blur_kernel<<<768, 256, 0, stream>>>(rank1, rank2, sigma, Bt, At);

    // 2) GEMM: 3 * 64 * 64 = 12288 wave tiles (32x32 each), 8 waves per block
    gemm_kernel<<<(3 * 64 * 64) / 8, 256, 0, stream>>>(At, Bt, img);

    // 3) bilinear gather
    const int total = NB * NPTS; // 2,097,152
    sample_kernel<<<(total + 255) / 256, 256, 0, stream>>>(img, coord, out, total);
}
